// MultiHeadAttention_3401614098690
// MI455X (gfx1250) — matrix operand order repaired and checked by reading, not compiled, not run on hardware
//
#include <hip/hip_runtime.h>

#define B_ 2
#define S_ 2048
#define D_ 1024
#define H_ 16
#define E_ 64
#define NT_ (S_ / 32)   // key tiles in flash loop

typedef __attribute__((ext_vector_type(16))) _Float16 h16;
typedef __attribute__((ext_vector_type(8)))  _Float16 h8;
typedef __attribute__((ext_vector_type(4)))  _Float16 h4;
typedef __attribute__((ext_vector_type(8)))  float    f8;

static __device__ __forceinline__ f8 wmma16(h16 a, h16 b, f8 c) {
  // D(16x16 f32) = A(16x32 f16) * B(32x16 f16) + C
  return __builtin_amdgcn_wmma_f32_16x16x32_f16(false, a, false, b, (short)0, c,
                                                false, false);
}

static __device__ __forceinline__ h16 cat(h8 lo, h8 hi) {
  h16 r;
#pragma unroll
  for (int i = 0; i < 8; ++i) { r[i] = lo[i]; r[i + 8] = hi[i]; }
  return r;
}

// Async global->LDS copy, 16B per lane, tracked by ASYNCcnt (CDNA5 TDM-class
// data path; no VGPR round trip).  LDS operand is the wave-relative byte
// offset = low 32 bits of the generic shared-pointer address.
static __device__ __forceinline__ void async_b128(const _Float16* g,
                                                  _Float16* l) {
  const unsigned lds = (unsigned)(unsigned long long)l;
  asm volatile("global_load_async_to_lds_b128 %0, %1, off"
               :: "v"(lds), "v"(g)
               : "memory");
}

// ---------------------------------------------------------------------------
// Kernel 1: per-head QKV projection.  Y[b,h,s,e] = sum_d X[b,s,d] * W[h,d,e]
// grid = (M/128, H, 3 {q,k,v}), block = 256 (8 waves x 16-row tiles).
// Q,K stored [B,H,S,E] f16; V stored transposed [B,H,E,S] f16.
// ---------------------------------------------------------------------------
__global__ __launch_bounds__(256) void k_qkv_proj(
    const float* __restrict__ q, const float* __restrict__ k,
    const float* __restrict__ v, const float* __restrict__ Wq,
    const float* __restrict__ Wk, const float* __restrict__ Wv,
    _Float16* __restrict__ Qh, _Float16* __restrict__ Kh,
    _Float16* __restrict__ VT) {
  const int which = blockIdx.z;
  const int h     = blockIdx.y;
  const int mT    = blockIdx.x;
  const float* X = (which == 0) ? q : (which == 1) ? k : v;
  const float* W = ((which == 0) ? Wq : (which == 1) ? Wk : Wv) +
                   (size_t)h * D_ * E_;

  __shared__ _Float16 Bt[64][40];  // W chunk, transposed [e][k], f16, padded

  const int tid  = threadIdx.x;
  const int wave = tid >> 5;
  const int lane = tid & 31;
  const int l16  = lane & 15;
  const int hs   = lane >> 4;

  const int rowBase = mT * 128 + wave * 16;
  const float* xr   = X + (size_t)(rowBase + l16) * D_;

  const int ecol = tid & 63;
  const int kg   = tid >> 6;

  f8 acc[4] = {};

  for (int k0 = 0; k0 < D_; k0 += 32) {
    __syncthreads();
    // Cooperative load of W[k0..k0+31][0..63] -> Bt[e][k] (f32->f16),
    // coalesced global reads (64 consecutive lanes sweep e).
#pragma unroll
    for (int hh = 0; hh < 2; ++hh) {
      const int kb = (kg + hh * 4) * 4;
      h4 t;
#pragma unroll
      for (int i = 0; i < 4; ++i)
        t[i] = (_Float16)W[(size_t)(k0 + kb + i) * E_ + ecol];
      *reinterpret_cast<h4*>(&Bt[ecol][kb]) = t;
    }
    __syncthreads();

    __builtin_prefetch(xr + k0 + 32, 0, 0);  // next A chunk -> global_prefetch

    // A fragment: row = l16, halves j<8 -> K=hs*8+j, j>=8 -> K=16+hs*8+(j-8)
    h16 a;
    {
      const float4 f0 = *reinterpret_cast<const float4*>(xr + k0 + hs * 8);
      const float4 f1 = *reinterpret_cast<const float4*>(xr + k0 + hs * 8 + 4);
      const float4 f2 = *reinterpret_cast<const float4*>(xr + k0 + 16 + hs * 8);
      const float4 f3 = *reinterpret_cast<const float4*>(xr + k0 + 20 + hs * 8);
      a[0] = (_Float16)f0.x; a[1] = (_Float16)f0.y;
      a[2] = (_Float16)f0.z; a[3] = (_Float16)f0.w;
      a[4] = (_Float16)f1.x; a[5] = (_Float16)f1.y;
      a[6] = (_Float16)f1.z; a[7] = (_Float16)f1.w;
      a[8]  = (_Float16)f2.x; a[9]  = (_Float16)f2.y;
      a[10] = (_Float16)f2.z; a[11] = (_Float16)f2.w;
      a[12] = (_Float16)f3.x; a[13] = (_Float16)f3.y;
      a[14] = (_Float16)f3.z; a[15] = (_Float16)f3.w;
    }
#pragma unroll
    for (int nt = 0; nt < 4; ++nt) {
      const _Float16* br = &Bt[nt * 16 + l16][hs * 8];
      h16 bb = cat(*reinterpret_cast<const h8*>(br),
                   *reinterpret_cast<const h8*>(br + 16));
      acc[nt] = wmma16(a, bb, acc[nt]);
    }
  }

  // Store.  C layout: VGPR r -> M = hs*8 + r, N = l16.
  // Uniform branch hoisted out of the store loops (no per-store branching).
  if (which == 2) {
#pragma unroll
    for (int nt = 0; nt < 4; ++nt)
#pragma unroll
      for (int r = 0; r < 8; ++r) {
        const int m  = rowBase + hs * 8 + r;
        const int bb = m >> 11;        // m / S_
        const int s  = m & (S_ - 1);
        const int e  = nt * 16 + l16;
        VT[(((size_t)bb * H_ + h) * E_ + e) * S_ + s] = (_Float16)acc[nt][r];
      }
  } else {
    _Float16* Y = (which == 0) ? Qh : Kh;
#pragma unroll
    for (int nt = 0; nt < 4; ++nt)
#pragma unroll
      for (int r = 0; r < 8; ++r) {
        const int m  = rowBase + hs * 8 + r;
        const int bb = m >> 11;
        const int s  = m & (S_ - 1);
        Y[(((size_t)bb * H_ + h) * S_ + s) * E_ + nt * 16 + l16] =
            (_Float16)acc[nt][r];
      }
  }
}

// ---------------------------------------------------------------------------
// Kernel 2: flash attention.  grid = (S/128, B*H), block = 256.
// Each wave owns 16 query rows; loops over 64 key tiles of 32 with online
// softmax.  K/V tiles double-buffered via async global->LDS copies.
// Writes multi-head output [B,S,D] f16.
// ---------------------------------------------------------------------------
__global__ __launch_bounds__(256) void k_flash(
    const _Float16* __restrict__ Qh, const _Float16* __restrict__ Kh,
    const _Float16* __restrict__ VT, const unsigned char* __restrict__ mask,
    _Float16* __restrict__ MH) {
  const int bh = blockIdx.y;
  const int b  = bh >> 4;
  const int h  = bh & 15;

  const int tid  = threadIdx.x;
  const int wave = tid >> 5;
  const int lane = tid & 31;
  const int l16  = lane & 15;
  const int hs   = lane >> 4;

  const _Float16* qb = Qh + (size_t)bh * S_ * E_;
  const _Float16* kb = Kh + (size_t)bh * S_ * E_;
  const _Float16* vb = VT + (size_t)bh * E_ * S_;
  const unsigned char* mb = mask + (size_t)b * S_ * S_;

  __shared__ _Float16 Klds[2][32][72];    // [buf][t][e]
  __shared__ _Float16 Vlds[2][64][40];    // [buf][e][t]
  __shared__ _Float16 Plds[8][16][40];    // per-wave P transpose slab

  const int qr0 = blockIdx.x * 128 + wave * 16;

  // Q A-fragments (K-dim = e, split 0..31 / 32..63), loaded once and
  // pre-scaled by 1/sqrt(E) = 0.125 (exact in f16).
  const _Float16* qrow = qb + (size_t)(qr0 + l16) * E_;
  h16 aQ0 = cat(*(const h8*)(qrow + hs * 8), *(const h8*)(qrow + 16 + hs * 8));
  h16 aQ1 = cat(*(const h8*)(qrow + 32 + hs * 8),
                *(const h8*)(qrow + 48 + hs * 8));
#pragma unroll
  for (int i = 0; i < 16; ++i) {
    aQ0[i] = aQ0[i] * (_Float16)0.125f;
    aQ1[i] = aQ1[i] * (_Float16)0.125f;
  }

  // Per-thread staging coordinates (one async b128 for K, one for V).
  const int skr = tid >> 3, skc = (tid & 7) * 8;   // K: 32 x 64 halfs
  const int svr = tid >> 2, svc = (tid & 3) * 8;   // V: 64 x 32 halfs

  auto stage = [&](int kt, int buf) {
    async_b128(kb + (size_t)(kt * 32 + skr) * E_ + skc, &Klds[buf][skr][skc]);
    async_b128(vb + (size_t)svr * S_ + kt * 32 + svc, &Vlds[buf][svr][svc]);
  };

  f8 O[4] = {};
  float mrow[8], lrow[8];
#pragma unroll
  for (int r = 0; r < 8; ++r) { mrow[r] = -1e30f; lrow[r] = 0.0f; }

  stage(0, 0);  // prologue: tile 0 in flight

  for (int kt = 0; kt < NT_; ++kt) {
    const int buf = kt & 1;
    if (kt + 1 < NT_) {
      stage(kt + 1, buf ^ 1);                       // overlap next tile's copy
      asm volatile("s_wait_asynccnt 0x2" ::: "memory");  // oldest 2 (cur) done
    } else {
      asm volatile("s_wait_asynccnt 0x0" ::: "memory");
    }
    __syncthreads();  // all waves' current-tile copies visible

    // scores: 16(q) x 32(keys), two 16-wide C tiles, K-dim = 64 (2 wmma each)
    f8 c01[2];
#pragma unroll
    for (int tt = 0; tt < 2; ++tt) {
      const _Float16* kr = &Klds[buf][tt * 16 + l16][0];
      h16 b0 = cat(*(const h8*)(kr + hs * 8), *(const h8*)(kr + 16 + hs * 8));
      h16 b1 = cat(*(const h8*)(kr + 32 + hs * 8),
                   *(const h8*)(kr + 48 + hs * 8));
      f8 z = {};
      z = wmma16(aQ0, b0, z);
      c01[tt] = wmma16(aQ1, b1, z);
    }

    // online softmax; each lane carries rows hs*8 + r, reductions stay
    // inside the 16-lane half (xor masks 1..8)
    float p0s[8], p1s[8], alpha[8];
#pragma unroll
    for (int r = 0; r < 8; ++r) {
      const int srow = qr0 + hs * 8 + r;
      const int tcol = kt * 32 + l16;
      float s0 = c01[0][r];
      float s1 = c01[1][r];
      s0 = mb[(size_t)srow * S_ + tcol]      ? s0 : -1e9f;
      s1 = mb[(size_t)srow * S_ + tcol + 16] ? s1 : -1e9f;
      float cm = fmaxf(s0, s1);
      cm = fmaxf(cm, __shfl_xor(cm, 1));
      cm = fmaxf(cm, __shfl_xor(cm, 2));
      cm = fmaxf(cm, __shfl_xor(cm, 4));
      cm = fmaxf(cm, __shfl_xor(cm, 8));
      const float mn = fmaxf(mrow[r], cm);
      const float al = __expf(mrow[r] - mn);
      const float p0 = __expf(s0 - mn);
      const float p1 = __expf(s1 - mn);
      float rs = p0 + p1;
      rs += __shfl_xor(rs, 1);
      rs += __shfl_xor(rs, 2);
      rs += __shfl_xor(rs, 4);
      rs += __shfl_xor(rs, 8);
      lrow[r] = lrow[r] * al + rs;
      mrow[r] = mn;
      alpha[r] = al; p0s[r] = p0; p1s[r] = p1;
    }
#pragma unroll
    for (int et = 0; et < 4; ++et)
#pragma unroll
      for (int r = 0; r < 8; ++r) O[et][r] *= alpha[r];

    // P: C layout -> A layout via wave-private LDS round trip
#pragma unroll
    for (int r = 0; r < 8; ++r) {
      Plds[wave][hs * 8 + r][l16]      = (_Float16)p0s[r];
      Plds[wave][hs * 8 + r][16 + l16] = (_Float16)p1s[r];
    }
    asm volatile("s_wait_dscnt 0x0" ::: "memory");
    const _Float16* pr = &Plds[wave][l16][0];
    const h16 aP = cat(*(const h8*)(pr + hs * 8),
                       *(const h8*)(pr + 16 + hs * 8));

    // O += P(16x32) * V(32x64): 4 e-tiles, K-dim = 32 keys
#pragma unroll
    for (int et = 0; et < 4; ++et) {
      const _Float16* vrp = &Vlds[buf][et * 16 + l16][hs * 8];
      h16 bV = cat(*(const h8*)(vrp), *(const h8*)(vrp + 16));
      O[et] = wmma16(aP, bV, O[et]);
    }
    __syncthreads();  // buf fully read before it is overwritten next iter
  }

#pragma unroll
  for (int et = 0; et < 4; ++et) {
#pragma unroll
    for (int r = 0; r < 8; ++r) {
      const int srow = qr0 + hs * 8 + r;
      const int e    = et * 16 + l16;
      MH[((size_t)b * S_ + srow) * D_ + h * E_ + e] =
          (_Float16)(O[et][r] / lrow[r]);
    }
  }
}

// ---------------------------------------------------------------------------
// Kernel 3: out = MH(f16)[4096,1024] @ Wo(f32)[1024,1024] -> f32
// grid = (M/128, N/64), block = 256.
// ---------------------------------------------------------------------------
__global__ __launch_bounds__(256) void k_out_proj(
    const _Float16* __restrict__ MH, const float* __restrict__ Wo,
    float* __restrict__ out) {
  const int mT = blockIdx.x;
  const int n0 = blockIdx.y * 64;
  __shared__ _Float16 Bt[64][40];  // Wo chunk transposed [n][k]

  const int tid  = threadIdx.x;
  const int wave = tid >> 5;
  const int lane = tid & 31;
  const int l16  = lane & 15;
  const int hs   = lane >> 4;

  const int rowBase   = mT * 128 + wave * 16;
  const _Float16* ar  = MH + (size_t)(rowBase + l16) * D_;

  const int ecol = tid & 63;
  const int kg   = tid >> 6;

  f8 acc[4] = {};

  for (int k0 = 0; k0 < D_; k0 += 32) {
    __syncthreads();
#pragma unroll
    for (int hh = 0; hh < 2; ++hh) {
      const int kb = (kg + hh * 4) * 4;
      h4 t;
#pragma unroll
      for (int i = 0; i < 4; ++i)
        t[i] = (_Float16)Wo[(size_t)(k0 + kb + i) * D_ + n0 + ecol];
      *reinterpret_cast<h4*>(&Bt[ecol][kb]) = t;
    }
    __syncthreads();

    h16 a = cat(*(const h8*)(ar + k0 + hs * 8),
                *(const h8*)(ar + k0 + 16 + hs * 8));
#pragma unroll
    for (int nt = 0; nt < 4; ++nt) {
      const _Float16* br = &Bt[nt * 16 + l16][hs * 8];
      h16 bb = cat(*(const h8*)br, *(const h8*)(br + 16));
      acc[nt] = wmma16(a, bb, acc[nt]);
    }
  }

#pragma unroll
  for (int nt = 0; nt < 4; ++nt)
#pragma unroll
    for (int r = 0; r < 8; ++r)
      out[(size_t)(rowBase + hs * 8 + r) * D_ + n0 + nt * 16 + l16] =
          acc[nt][r];
}

// ---------------------------------------------------------------------------
extern "C" void kernel_launch(void* const* d_in, const int* in_sizes, int n_in,
                              void* d_out, int out_size, void* d_ws,
                              size_t ws_size, hipStream_t stream) {
  (void)in_sizes; (void)n_in; (void)out_size; (void)ws_size;
  const float* q  = (const float*)d_in[0];
  const float* k  = (const float*)d_in[1];
  const float* v  = (const float*)d_in[2];
  const unsigned char* mask = (const unsigned char*)d_in[3];
  const float* Wq = (const float*)d_in[4];
  const float* Wk = (const float*)d_in[5];
  const float* Wv = (const float*)d_in[6];
  const float* Wo = (const float*)d_in[7];
  float* out = (float*)d_out;

  char* ws = (char*)d_ws;
  const size_t MB8 = (size_t)8 * 1024 * 1024;
  _Float16* Qh = (_Float16*)(ws);             // [B,H,S,E] f16
  _Float16* Kh = (_Float16*)(ws + MB8);       // [B,H,S,E] f16
  _Float16* VT = (_Float16*)(ws + 2 * MB8);   // [B,H,E,S] f16
  _Float16* MH = (_Float16*)(ws + 3 * MB8);   // [B,S,D]   f16

  k_qkv_proj<<<dim3((B_ * S_) / 128, H_, 3), 256, 0, stream>>>(
      q, k, v, Wq, Wk, Wv, Qh, Kh, VT);
  k_flash<<<dim3(S_ / 128, B_ * H_), 256, 0, stream>>>(Qh, Kh, VT, mask, MH);
  k_out_proj<<<dim3((B_ * S_) / 128, D_ / 64), 256, 0, stream>>>(MH, Wo, out);
}
